// KATFModel_19026705121690
// MI455X (gfx1250) — compile-verified, matrix operand order checked by reading, not run on hardware
//
#include <hip/hip_runtime.h>
#include <hip/hip_bf16.h>
#include <math.h>

// ---------------- types ----------------
typedef __attribute__((ext_vector_type(16))) _Float16 v16h;
typedef __attribute__((ext_vector_type(8)))  _Float16 v8h;
typedef __attribute__((ext_vector_type(8)))  float    v8f;

#define BN   256
#define TN   65
#define HIDN 256
#define RROWS (BN*TN)   // 16640 = 520 * 32

// ---------------- WMMA fragment loader ----------------
// 16-bit A/B fragment for V_WMMA_F32_16X16X32_F16 (wave32).
// lane<16 : row = lane, halves 0..7 = K 0..7,  halves 8..15 = K 16..23
// lane>=16: row = lane-16, halves 0..7 = K 8..15, halves 8..15 = K 24..31
__device__ inline v16h load_frag(const _Float16* rowptr, int khalf) {
    union { v16h v; v8h h[2]; } u;
    u.h[0] = *(const v8h*)(rowptr + khalf);
    u.h[1] = *(const v8h*)(rowptr + khalf + 16);
    return u.v;
}

__device__ inline float sigm(float x) { return 1.0f / (1.0f + expf(-x)); }

// ---------------- conv1: direct fp32 (Cin*K = 150, too small for WMMA) ----------------
// input x: (B, Cin, Lin) channel-major; output y: (B, Lout, Cout)
__global__ void conv1_direct(const float* __restrict__ x, const float* __restrict__ W,
                             const float* __restrict__ bias, const float* __restrict__ gamma,
                             const float* __restrict__ beta, const float* __restrict__ mean,
                             const float* __restrict__ var, float* __restrict__ y,
                             int Cin, int Lin, int Cout, int Lout, int K, int stride, int pad,
                             int total) {
    int idx = blockIdx.x * blockDim.x + threadIdx.x;
    if (idx >= total) return;
    int l  = idx % Lout;
    int co = (idx / Lout) % Cout;
    int b  = idx / (Lout * Cout);
    const float* xb = x + (size_t)b * Cin * Lin;
    const float* wc = W + (size_t)co * Cin * K;
    int l0 = l * stride - pad;
    float acc = 0.f;
    for (int ci = 0; ci < Cin; ++ci) {
        const float* xr = xb + (size_t)ci * Lin;
        const float* wr = wc + (size_t)ci * K;
        for (int kk = 0; kk < K; ++kk) {
            int li = l0 + kk;
            if (li >= 0 && li < Lin) acc += xr[li] * wr[kk];
        }
    }
    float scale = gamma[co] * rsqrtf(var[co] + 1e-5f);
    float v = (acc + bias[co] - mean[co]) * scale + beta[co];
    y[((size_t)b * Lout + l) * Cout + co] = fmaxf(v, 0.f);
}

// ---------------- im2col (f32 (B,Lin,Cin) -> f16 (B*Lout, Kp)), zero-padded K ----------------
__global__ void im2col_f16(const float* __restrict__ in, _Float16* __restrict__ out,
                           int Cin, int Lin, int Lout, int K, int stride, int pad, int Kp,
                           int total) {
    int idx = blockIdx.x * blockDim.x + threadIdx.x;
    if (idx >= total) return;
    int q = idx % Kp;
    int l = (idx / Kp) % Lout;
    int b = idx / (Kp * Lout);
    float v = 0.f;
    if (q < Cin * K) {
        int ci = q / K;
        int kk = q % K;
        int li = l * stride - pad + kk;
        if (li >= 0 && li < Lin) v = in[((size_t)b * Lin + li) * Cin + ci];
    }
    out[idx] = (_Float16)v;
}

// (Cout, Kc) f32 -> (Cout, Kp) f16, zero-padded
__global__ void pad_w_f16(const float* __restrict__ W, _Float16* __restrict__ out,
                          int Kc, int Kp, int total) {
    int idx = blockIdx.x * blockDim.x + threadIdx.x;
    if (idx >= total) return;
    int q  = idx % Kp;
    int co = idx / Kp;
    out[idx] = (_Float16)((q < Kc) ? W[(size_t)co * Kc + q] : 0.f);
}

// fold conv bias + BN into scale/shift: v = acc*scale[c] + shift[c]
__global__ void bn_fold(const float* __restrict__ b, const float* __restrict__ g,
                        const float* __restrict__ be, const float* __restrict__ mu,
                        const float* __restrict__ var, float* __restrict__ scale,
                        float* __restrict__ shift, int C) {
    int c = blockIdx.x * blockDim.x + threadIdx.x;
    if (c >= C) return;
    float s = g[c] * rsqrtf(var[c] + 1e-5f);
    scale[c] = s;
    shift[c] = (b[c] - mu[c]) * s + be[c];
}

__global__ void f32_to_f16(const float* __restrict__ in, _Float16* __restrict__ out, int n) {
    int i = blockIdx.x * blockDim.x + threadIdx.x;
    if (i < n) out[i] = (_Float16)in[i];
}

__global__ void zero_f32(float* __restrict__ p, int n) {
    int i = blockIdx.x * blockDim.x + threadIdx.x;
    if (i < n) p[i] = 0.f;
}

// ---------------- WMMA GEMM: Y(R,M) = A16(R,K) * Bw16(M,K)^T, fused epilogue ----------------
// block = 128 threads (4 waves). Each wave: 32 rows x 64 cols (8 accumulators, B frags
// shared across both row halves -> 1.5 b128 loads per wmma). grid = (ceil(M/256), R/32).
// mode 0: Y = acc + bias0 (+bias1)          (LSTM gate projection, biases folded)
// mode 1: Y = softplus(acc + bias0)         (Kalman noise)
// mode 2: Y = relu(acc*bias0 + bias1)       (conv BN+ReLU; bias0=scale, bias1=shift)
__global__ void wmma_gemm(const _Float16* __restrict__ A, const _Float16* __restrict__ Bw,
                          const float* __restrict__ bias0, const float* __restrict__ bias1,
                          float* __restrict__ Y, int R, int M, int K, int mode) {
    int lane  = threadIdx.x & 31;
    int wave  = threadIdx.x >> 5;
    int row0  = blockIdx.y * 32;
    int col0  = blockIdx.x * 256 + wave * 64;
    int lm    = lane & 15;
    int khalf = (lane >= 16) ? 8 : 0;

    const _Float16* ar0 = A + (size_t)(row0 + lm) * K;
    const _Float16* ar1 = A + (size_t)(row0 + 16 + lm) * K;
    const _Float16* br[4];
#pragma unroll
    for (int q = 0; q < 4; ++q) {
        int bc = col0 + q * 16 + lm;
        if (bc > M - 1) bc = M - 1;  // clamp for M not multiple of 256 (safe loads)
        br[q] = Bw + (size_t)bc * K;
    }

    v8f acc[2][4] = {};
    for (int k = 0; k < K; k += 32) {
        __builtin_prefetch(ar0 + k + 128, 0, 1);
        __builtin_prefetch(ar1 + k + 128, 0, 1);
        v16h a0 = load_frag(ar0 + k, khalf);
        v16h a1 = load_frag(ar1 + k, khalf);
#pragma unroll
        for (int q = 0; q < 4; ++q) {
            v16h bf = load_frag(br[q] + k, khalf);
            acc[0][q] = __builtin_amdgcn_wmma_f32_16x16x32_f16(false, a0, false, bf, (short)0, acc[0][q], false, false);
            acc[1][q] = __builtin_amdgcn_wmma_f32_16x16x32_f16(false, a1, false, bf, (short)0, acc[1][q], false, false);
        }
    }

    int hi8 = (lane >= 16) ? 8 : 0;
#pragma unroll
    for (int rh = 0; rh < 2; ++rh) {
        int rbase = row0 + rh * 16 + hi8;
#pragma unroll
        for (int q = 0; q < 4; ++q) {
            int col = col0 + q * 16 + lm;
            if (col >= M) continue;
            float b0 = bias0 ? bias0[col] : 0.f;
            float b1 = bias1 ? bias1[col] : 0.f;
#pragma unroll
            for (int j = 0; j < 8; ++j) {
                float a = acc[rh][q][j];
                float v;
                if (mode == 2) {
                    v = fmaxf(a * b0 + b1, 0.f);
                } else if (mode == 1) {
                    float t = a + b0;
                    v = fmaxf(t, 0.f) + log1pf(expf(-fabsf(t)));  // stable softplus
                } else {
                    v = a + b0 + b1;
                }
                Y[(size_t)(rbase + j) * M + col] = v;
            }
        }
    }
}

// ---------------- fused LSTM recurrent step ----------------
// One wave per 16 batch x 16 hidden tile; 4 WMMA accumulators = i,f,g,o gates.
// grid = (16 batch tiles, 16 hidden tiles), block = 32.
__global__ void lstm_step(const float* __restrict__ xg,       // (B,T,1024), biases folded in
                          const _Float16* __restrict__ Whh16, // (1024,256)
                          const _Float16* __restrict__ hin,   // (B,256) f16 (read)
                          _Float16* __restrict__ hout,        // (B,256) f16 (write)
                          float* __restrict__ c32,            // (B,256)
                          float* __restrict__ hs,             // (B,T,512)
                          int t, int dirofs, int T) {
    int lane  = threadIdx.x & 31;
    int bt    = blockIdx.x;
    int ht    = blockIdx.y;
    int lm    = lane & 15;
    int khalf = (lane >= 16) ? 8 : 0;

    const _Float16* arow = hin + (size_t)(bt * 16 + lm) * HIDN;
    const _Float16* bri = Whh16 + (size_t)(0 * 256 + ht * 16 + lm) * HIDN;
    const _Float16* brf = Whh16 + (size_t)(1 * 256 + ht * 16 + lm) * HIDN;
    const _Float16* brg = Whh16 + (size_t)(2 * 256 + ht * 16 + lm) * HIDN;
    const _Float16* bro = Whh16 + (size_t)(3 * 256 + ht * 16 + lm) * HIDN;

    v8f ai = {}, af_ = {}, ag = {}, ao = {};
    for (int k = 0; k < HIDN; k += 32) {
        v16h afrag = load_frag(arow + k, khalf);
        ai  = __builtin_amdgcn_wmma_f32_16x16x32_f16(false, afrag, false, load_frag(bri + k, khalf), (short)0, ai,  false, false);
        af_ = __builtin_amdgcn_wmma_f32_16x16x32_f16(false, afrag, false, load_frag(brf + k, khalf), (short)0, af_, false, false);
        ag  = __builtin_amdgcn_wmma_f32_16x16x32_f16(false, afrag, false, load_frag(brg + k, khalf), (short)0, ag,  false, false);
        ao  = __builtin_amdgcn_wmma_f32_16x16x32_f16(false, afrag, false, load_frag(bro + k, khalf), (short)0, ao,  false, false);
    }
    int hi8 = (lane >= 16) ? 8 : 0;
    int hid = ht * 16 + lm;
    for (int j = 0; j < 8; ++j) {
        int b = bt * 16 + j + hi8;
        size_t xb = ((size_t)b * T + t) * 1024;
        float gi = ai[j]  + xg[xb + 0   + hid];
        float gf = af_[j] + xg[xb + 256 + hid];
        float gg = ag[j]  + xg[xb + 512 + hid];
        float go = ao[j]  + xg[xb + 768 + hid];
        size_t ci = (size_t)b * HIDN + hid;
        float cn = sigm(gf) * c32[ci] + sigm(gi) * tanhf(gg);
        float hn = sigm(go) * tanhf(cn);
        c32[ci]  = cn;
        hout[ci] = (_Float16)hn;
        hs[((size_t)b * T + t) * 512 + dirofs + hid] = hn;
    }
}

// ---------------- Kalman scan (sequential over T per (b,j)) ----------------
__global__ void kalman_scan(const float* __restrict__ h, const float* __restrict__ noise,
                            float* __restrict__ out, int T) {
    int idx = blockIdx.x * blockDim.x + threadIdx.x;
    if (idx >= BN * 512) return;
    int j = idx & 511;
    int b = idx >> 9;
    float s = h[((size_t)b * T) * 512 + j];
    float P = 0.1f;
    for (int t = 0; t < T; ++t) {
        size_t o = (size_t)b * T + t;
        float z = h[o * 512 + j];
        float q = noise[o * 1024 + j];
        float r = noise[o * 1024 + 512 + j];
        float Pp = P + q;
        float Kg = Pp / (Pp + r + 1e-6f);
        s = s + Kg * (z - s);
        P = (1.f - Kg) * Pp;
        out[o * 512 + j] = s;
    }
}

// ---------------- attention scores: block per (b,t), 64 threads ----------------
__global__ void attn_scores(const float* __restrict__ h, const float* __restrict__ W1,
                            const float* __restrict__ b1, const float* __restrict__ W2,
                            const float* __restrict__ b2, float* __restrict__ scores) {
    __shared__ float red[64];
    int bt = blockIdx.x;
    int u  = threadIdx.x;
    const float* hv = h + (size_t)bt * 512;
    const float* w  = W1 + (size_t)u * 512;
    float acc = b1[u];
    for (int d = 0; d < 512; ++d) acc += hv[d] * w[d];
    red[u] = tanhf(acc) * W2[u];
    __syncthreads();
    for (int s = 32; s > 0; s >>= 1) {
        if (u < s) red[u] += red[u + s];
        __syncthreads();
    }
    if (u == 0) scores[bt] = red[0] + b2[0];
}

// ---------------- softmax over T + context pooling: block per b ----------------
__global__ void attn_ctx(const float* __restrict__ h, const float* __restrict__ scores,
                         float* __restrict__ ctx, int T) {
    __shared__ float w[TN];
    __shared__ float mx, sm;
    int b = blockIdx.x;
    int tid = threadIdx.x;  // 128
    if (tid < T) w[tid] = scores[b * T + tid];
    __syncthreads();
    if (tid == 0) {
        float m = -1e30f;
        for (int t = 0; t < T; ++t) m = fmaxf(m, w[t]);
        mx = m;
    }
    __syncthreads();
    if (tid < T) w[tid] = expf(w[tid] - mx);
    __syncthreads();
    if (tid == 0) {
        float s = 0.f;
        for (int t = 0; t < T; ++t) s += w[t];
        sm = s;
    }
    __syncthreads();
    for (int d = tid; d < 512; d += 128) {
        float a = 0.f;
        for (int t = 0; t < T; ++t) a += w[t] * h[((size_t)b * T + t) * 512 + d];
        ctx[(size_t)b * 512 + d] = a / sm;
    }
}

// ---------------- classifier ----------------
__global__ void clf1(const float* __restrict__ ctx, const float* __restrict__ W1,
                     const float* __restrict__ b1, const float* __restrict__ gamma,
                     const float* __restrict__ beta, const float* __restrict__ mean,
                     const float* __restrict__ var, float* __restrict__ y1) {
    int idx = blockIdx.x * blockDim.x + threadIdx.x;
    if (idx >= BN * 256) return;
    int m = idx & 255;
    int b = idx >> 8;
    const float* cv = ctx + (size_t)b * 512;
    const float* wv = W1 + (size_t)m * 512;
    float acc = b1[m];
    for (int d = 0; d < 512; ++d) acc += cv[d] * wv[d];
    float scale = gamma[m] * rsqrtf(var[m] + 1e-5f);
    float v = (acc - mean[m]) * scale + beta[m];
    y1[idx] = fmaxf(v, 0.f);
}

__global__ void clf2(const float* __restrict__ y1, const float* __restrict__ W2,
                     const float* __restrict__ b2, float* __restrict__ out) {
    int idx = blockIdx.x * blockDim.x + threadIdx.x;
    if (idx >= BN * 4) return;
    int n = idx & 3;
    int b = idx >> 2;
    const float* yv = y1 + (size_t)b * 256;
    const float* wv = W2 + (size_t)n * 256;
    float acc = b2[n];
    for (int m = 0; m < 256; ++m) acc += yv[m] * wv[m];
    out[idx] = acc;
}

// ---------------- host-side orchestration ----------------
#define MB ((size_t)1 << 20)
static const size_t OFF_A      = 0;        // conv activations ping (32MB)
static const size_t OFF_B      = 32 * MB;  // conv activations pong (32MB)
static const size_t OFF_X16    = 64 * MB;  // f16 GEMM input (<=17MB)
static const size_t OFF_XG0    = 84 * MB;  // im2col / xg fwd / kalman noise (<=68.2MB)
static const size_t OFF_XG1    = 156 * MB; // xg bwd / kalman out (68.2MB)
static const size_t OFF_HS     = 228 * MB; // layer output (34.1MB)
static const size_t OFF_H16A   = 264 * MB;
static const size_t OFF_H16B   = 265 * MB;
static const size_t OFF_C32    = 266 * MB;
static const size_t OFF_W16    = 267 * MB; // f16 weights (<=2MB)
static const size_t OFF_WHH16  = 269 * MB; // f16 Whh (512KB)
static const size_t OFF_SCORES = 270 * MB;
static const size_t OFF_CTX    = 271 * MB;
static const size_t OFF_Y1     = 272 * MB;
static const size_t OFF_BN0    = 273 * MB; // folded BN scale
static const size_t OFF_BN1    = OFF_BN0 + 4096;

extern "C" void kernel_launch(void* const* d_in, const int* in_sizes, int n_in,
                              void* d_out, int out_size, void* d_ws, size_t ws_size,
                              hipStream_t stream) {
    (void)in_sizes; (void)n_in; (void)out_size; (void)ws_size;
    auto FIN = [&](int i) { return (const float*)d_in[i]; };
    auto WSF = [&](size_t off) { return (float*)((char*)d_ws + off); };
    auto WSH = [&](size_t off) { return (_Float16*)((char*)d_ws + off); };

    // ---- conv1: direct ----
    {
        int total = BN * 48 * 390;
        conv1_direct<<<(total + 255) / 256, 256, 0, stream>>>(
            FIN(0), FIN(1), FIN(2), FIN(3), FIN(4), FIN(5), FIN(6),
            WSF(OFF_A), 6, 1950, 48, 390, 25, 5, 12, total);
    }

    // ---- conv2..4: im2col + WMMA GEMM with fused BN+ReLU ----
    struct CC { int cin, lin, cout, lout, k, st, pd, kp; };
    const CC cc[3] = {
        {48,  390, 96,  130, 15, 3, 7, 736},  // Kc=720 -> pad 736
        {96,  130, 192, 65,  7,  2, 3, 672},  // Kc=672
        {192, 65,  384, 65,  5,  1, 2, 960},  // Kc=960
    };
    float* act[2] = {WSF(OFF_A), WSF(OFF_B)};
    _Float16* colbuf = WSH(OFF_XG0);
    _Float16* w16    = WSH(OFF_W16);
    for (int i = 0; i < 3; ++i) {
        int base = 1 + 6 * (i + 1);  // conv params: W,b,gamma,beta,mean,var
        const float* srcA = act[i & 1];
        float* dstA = act[(i + 1) & 1];
        int Rr = BN * cc[i].lout;
        int Kc = cc[i].cin * cc[i].k;
        int tot_col = Rr * cc[i].kp;
        im2col_f16<<<(tot_col + 255) / 256, 256, 0, stream>>>(
            srcA, colbuf, cc[i].cin, cc[i].lin, cc[i].lout, cc[i].k, cc[i].st, cc[i].pd,
            cc[i].kp, tot_col);
        int tot_w = cc[i].cout * cc[i].kp;
        pad_w_f16<<<(tot_w + 255) / 256, 256, 0, stream>>>(FIN(base), w16, Kc, cc[i].kp, tot_w);
        bn_fold<<<(cc[i].cout + 63) / 64, 64, 0, stream>>>(
            FIN(base + 1), FIN(base + 2), FIN(base + 3), FIN(base + 4), FIN(base + 5),
            WSF(OFF_BN0), WSF(OFF_BN1), cc[i].cout);
        wmma_gemm<<<dim3((cc[i].cout + 255) / 256, Rr / 32), 128, 0, stream>>>(
            colbuf, w16, WSF(OFF_BN0), WSF(OFF_BN1), dstA, Rr, cc[i].cout, cc[i].kp, 2);
    }

    // conv4 out is (B, T, 384) f32 at OFF_B -> f16 LSTM input, no transpose needed
    {
        int n = RROWS * 384;
        f32_to_f16<<<(n + 255) / 256, 256, 0, stream>>>(WSF(OFF_B), WSH(OFF_X16), n);
    }

    // ---- bidirectional LSTM, 3 layers ----
    _Float16* x16   = WSH(OFF_X16);
    _Float16* whh16 = WSH(OFF_WHH16);
    float*    xgbuf[2] = {WSF(OFF_XG0), WSF(OFF_XG1)};
    float*    hsbuf = WSF(OFF_HS);
    float*    c32p  = WSF(OFF_C32);

    for (int l = 0; l < 3; ++l) {
        int Kl = (l == 0) ? 384 : 512;
        // input projections (both directions), biases bih+bhh folded
        for (int dir = 0; dir < 2; ++dir) {
            int pb = 25 + 8 * l + 4 * dir;  // Wih, Whh, bih, bhh
            int nw = 1024 * Kl;
            f32_to_f16<<<(nw + 255) / 256, 256, 0, stream>>>(FIN(pb), w16, nw);
            wmma_gemm<<<dim3(4, RROWS / 32), 128, 0, stream>>>(
                x16, w16, FIN(pb + 2), FIN(pb + 3), xgbuf[dir], RROWS, 1024, Kl, 0);
        }
        // recurrences
        for (int dir = 0; dir < 2; ++dir) {
            int pb = 25 + 8 * l + 4 * dir;
            int nw = 1024 * HIDN;
            f32_to_f16<<<(nw + 255) / 256, 256, 0, stream>>>(FIN(pb + 1), whh16, nw);
            zero_f32<<<(32768 + 255) / 256, 256, 0, stream>>>((float*)WSH(OFF_H16A), 32768);
            zero_f32<<<(65536 + 255) / 256, 256, 0, stream>>>(c32p, 65536);
            _Float16* hp = WSH(OFF_H16A);
            _Float16* hq = WSH(OFF_H16B);
            for (int s = 0; s < TN; ++s) {
                int t = dir ? (TN - 1 - s) : s;
                lstm_step<<<dim3(16, 16), 32, 0, stream>>>(
                    xgbuf[dir], whh16, hp, hq, c32p, hsbuf, t, dir * 256, TN);
                _Float16* tmp = hp; hp = hq; hq = tmp;
            }
        }
        if (l < 2) {
            int n = RROWS * 512;
            f32_to_f16<<<(n + 255) / 256, 256, 0, stream>>>(hsbuf, x16, n);
        }
    }

    // ---- Kalman: noise = softplus(h @ W^T + b), then scan ----
    {
        int n = RROWS * 512;
        f32_to_f16<<<(n + 255) / 256, 256, 0, stream>>>(hsbuf, x16, n);
        int nw = 1024 * 512;
        f32_to_f16<<<(nw + 255) / 256, 256, 0, stream>>>(FIN(49), w16, nw);
        wmma_gemm<<<dim3(4, RROWS / 32), 128, 0, stream>>>(
            x16, w16, FIN(50), nullptr, WSF(OFF_XG0), RROWS, 1024, 512, 1);
        int nth = BN * 512;
        kalman_scan<<<(nth + 255) / 256, 256, 0, stream>>>(hsbuf, WSF(OFF_XG0), WSF(OFF_XG1), TN);
    }

    // ---- attention + classifier ----
    float* kout = WSF(OFF_XG1);
    attn_scores<<<RROWS, 64, 0, stream>>>(kout, FIN(51), FIN(52), FIN(53), FIN(54), WSF(OFF_SCORES));
    attn_ctx<<<BN, 128, 0, stream>>>(kout, WSF(OFF_SCORES), WSF(OFF_CTX), TN);
    {
        int n1 = BN * 256;
        clf1<<<(n1 + 255) / 256, 256, 0, stream>>>(WSF(OFF_CTX), FIN(55), FIN(56), FIN(57), FIN(58),
                                                   FIN(59), FIN(60), WSF(OFF_Y1));
        int n2 = BN * 4;
        clf2<<<(n2 + 63) / 64, 64, 0, stream>>>(WSF(OFF_Y1), FIN(61), FIN(62), (float*)d_out);
    }
}
